// LlamaFlashAttention_KIVI_Opt_76622216561081
// MI455X (gfx1250) — matrix operand order repaired and checked by reading, not compiled, not run on hardware
//
#include <hip/hip_runtime.h>
#include <hip/hip_bf16.h>
#include <math.h>

typedef float  v2f   __attribute__((ext_vector_type(2)));
typedef float  v4f   __attribute__((ext_vector_type(4)));
typedef float  v8f   __attribute__((ext_vector_type(8)));
typedef int    v4i   __attribute__((ext_vector_type(4)));
typedef __bf16 v16bf __attribute__((ext_vector_type(16)));

#define HD     128
#define NH     32
#define NKV    8
#define GQA    4
#define LQ     3968
#define LR     128
#define NB     16
#define HID    4096
#define QKV_LD 6144
#define ATT_SCALE 0.088388347648318447f   /* 1/sqrt(128) */
#define ROPE_THETA 500000.0f
#define NCH    32
#define PSTR   520                        /* 4*128 acc + 4 M + 4 S */

__device__ __forceinline__ v4f ntld4f(const float* p) {
  return __builtin_nontemporal_load((const v4f*)p);
}
__device__ __forceinline__ v4i ntld4i(const int* p) {
  return __builtin_nontemporal_load((const v4i*)p);
}

// ---------------------------------------------------------------------------
// Shared GEMM body: Y[batch 0..15][out] = sum_k X[batch][k] * W[out][k].
// f32 WMMA 16x16x4; 8 waves split K; per 8-k iteration one float4 per lane
// feeds TWO wmmas via the slot-permutation trick (K-sum is order invariant:
// slots {k,k+1,k+4,k+5} then {k+2,k+3,k+6,k+7}, identical mapping on A & B).
// ---------------------------------------------------------------------------
__device__ __forceinline__ void gemm16_body(
    const float* __restrict__ X, const float* __restrict__ W, int K,
    float* __restrict__ Y, int ldy, int wrow0, int ycol0)
{
  __shared__ float red[8][16][16];
  const int lane = threadIdx.x & 31;
  const int wave = threadIdx.x >> 5;
  const int col  = lane & 15;
  const int hi   = lane >> 4;
  const int kcnt = K >> 3;                       // K per wave
  const int kbeg = wave * kcnt;
  const float* Xb = X + (size_t)col * K + kbeg + hi * 4;            // batch row
  const float* Wb = W + ((size_t)(wrow0 + col)) * K + kbeg + hi * 4; // out row
  v8f c = {0.f,0.f,0.f,0.f,0.f,0.f,0.f,0.f};
  #pragma unroll 8
  for (int it = 0; it < kcnt / 8; ++it) {
    v4f a4 = *(const v4f*)(Xb + it * 8);          // X reused by all blocks: temporal
    v4f b4 = ntld4f(Wb + it * 8);                 // W streamed once: non-temporal
    v2f a0, a1, b0, b1;
    a0.x = a4.x; a0.y = a4.y;  a1.x = a4.z; a1.y = a4.w;
    b0.x = b4.x; b0.y = b4.y;  b1.x = b4.z; b1.y = b4.w;
    c = __builtin_amdgcn_wmma_f32_16x16x4_f32(false, a0, false, b0, (short)0, c, false, false);
    c = __builtin_amdgcn_wmma_f32_16x16x4_f32(false, a1, false, b1, (short)0, c, false, false);
  }
  #pragma unroll
  for (int r = 0; r < 8; ++r) red[wave][hi * 8 + r][col] = c[r];
  __syncthreads();
  const int mr = threadIdx.x >> 4;   // batch row 0..15
  const int nc = threadIdx.x & 15;   // out col 0..15
  float s = 0.f;
  #pragma unroll
  for (int w = 0; w < 8; ++w) s += red[w][mr][nc];
  Y[(size_t)mr * ldy + ycol0 + nc] = s;
}

// Fused Q/K/V projections: 256 + 64 + 64 = 384 tiles in one launch.
__global__ __launch_bounds__(256) void gemm_qkv(
    const float* __restrict__ X, const float* __restrict__ Wq,
    const float* __restrict__ Wk, const float* __restrict__ Wv,
    float* __restrict__ Y)
{
  const int t = blockIdx.x;
  const float* W; int wrow0, ycol0;
  if (t < 256)      { W = Wq; wrow0 = t * 16;         ycol0 = wrow0; }
  else if (t < 320) { W = Wk; wrow0 = (t - 256) * 16; ycol0 = HID + wrow0; }
  else              { W = Wv; wrow0 = (t - 320) * 16; ycol0 = HID + 1024 + wrow0; }
  gemm16_body(X, W, HID, Y, QKV_LD, wrow0, ycol0);
}

// Output projection.
__global__ __launch_bounds__(256) void gemm_out(
    const float* __restrict__ X, const float* __restrict__ W,
    float* __restrict__ Y)
{
  gemm16_body(X, W, HID, Y, HID, blockIdx.x * 16, blockIdx.x * 16);
}

// ---------------------------------------------------------------------------
// RoPE on q (with softmax SCALE folded in) and k; v untouched in qkv buffer.
// ---------------------------------------------------------------------------
__global__ __launch_bounds__(256) void rope_qk(
    const float* __restrict__ qkv, const int* __restrict__ pos_ids,
    float* __restrict__ q_r, float* __restrict__ k_r)
{
  int idx = blockIdx.x * 256 + threadIdx.x;
  if (idx >= NB * (NH + NKV) * HD) return;
  int d = idx & (HD - 1);
  int h = (idx >> 7) % (NH + NKV);
  int b = idx / ((NH + NKV) * HD);
  bool isq = h < NH;
  const float* src = qkv + (size_t)b * QKV_LD + (isq ? h * HD : HID + (h - NH) * HD);
  float x  = src[d];
  float xp = (d < 64) ? -src[d + 64] : src[d - 64];
  float pos = (float)pos_ids[b];
  int j = d & 63;
  float inv = powf(ROPE_THETA, -(float)(2 * j) / 128.0f);
  float sn, cs;
  __sincosf(pos * inv, &sn, &cs);
  float y = x * cs + xp * sn;
  if (isq) q_r[((size_t)b * NH + h) * HD + d] = y * ATT_SCALE;
  else     k_r[((size_t)b * NKV + (h - NH)) * HD + d] = y;
}

// ---------------------------------------------------------------------------
// Flash-attention partials: one wave per (chunk, kv, b). Chunk = 128 keys.
// Chunks 0..30: quantized cache (31*128 = 3968). Chunk 31: full cache (128).
// Scores computed transposed: C = K * Q^T so that the C layout (lane=head,
// VGPR r = key hi*8+r) IS the A-fragment layout for P*V -> no shuffles.
// ---------------------------------------------------------------------------
__global__ __launch_bounds__(32) void attn_partial(
    const float* __restrict__ q_r,
    const int*   __restrict__ kq, const float* __restrict__ ks, const float* __restrict__ km,
    const int*   __restrict__ vq, const float* __restrict__ vs, const float* __restrict__ vm,
    const float* __restrict__ kfull, const float* __restrict__ vfull,
    float* __restrict__ partial)
{
  const int c    = blockIdx.x;
  const int kv   = blockIdx.y;
  const int b    = blockIdx.z;
  const int lane = threadIdx.x & 31;
  const int col  = lane & 15;
  const int hi   = lane >> 4;
  const int bkv  = b * NKV + kv;
  const bool quant = (c < NCH - 1);

  // Q^T B-fragments: lane col = head (0..3 real, rest zero). elem e: d = s*32+hi*8+e+(e&8)
  v16bf qt[4];
  {
    const int hcl = (col < GQA) ? col : 0;
    const float* qp = q_r + ((size_t)b * NH + kv * GQA + hcl) * HD;
    #pragma unroll
    for (int s = 0; s < 4; ++s) {
      v16bf t;
      #pragma unroll
      for (int e = 0; e < 16; ++e) {
        float v = (col < GQA) ? qp[s * 32 + hi * 8 + e + (e & 8)] : 0.0f;
        t[e] = (__bf16)v;
      }
      qt[s] = t;
    }
  }

  float M0 = -1e30f, M1 = -1e30f, M2 = -1e30f, M3 = -1e30f;
  float S0 = 0.f, S1 = 0.f, S2 = 0.f, S3 = 0.f;
  v8f acc[8];
  #pragma unroll
  for (int dt = 0; dt < 8; ++dt) acc[dt] = (v8f){0.f,0.f,0.f,0.f,0.f,0.f,0.f,0.f};

  for (int kb = 0; kb < 4; ++kb) {
    const int base = c * 128 + kb * 32;

    // lane j owns key base+j for the V scale/min tables (gathered by shuffle later)
    v4f vs4 = {1.f,1.f,1.f,1.f}, vm4 = {0.f,0.f,0.f,0.f};
    if (quant) {
      const size_t rowj = (size_t)bkv * LQ + base + lane;
      vs4 = ntld4f(vs + rowj * 4);
      vm4 = ntld4f(vm + rowj * 4);
    }

    v8f sc[2];
    #pragma unroll
    for (int t = 0; t < 2; ++t) {
      v8f s8 = {0.f,0.f,0.f,0.f,0.f,0.f,0.f,0.f};
      const int key = base + t * 16 + col;         // lane's key (A-matrix row)
      if (quant) {
        const int*   cp = kq + ((size_t)bkv * LQ + key) * HD;
        const v4f ks4 = ntld4f(ks + ((size_t)bkv * LQ + key) * 4);
        const v4f km4 = ntld4f(km + ((size_t)bkv * LQ + key) * 4);
        #pragma unroll
        for (int s = 0; s < 4; ++s) {
          const float scl = ks4[s], mn = km4[s];
          const int* r0 = cp + s * 32 + hi * 8;    // d = s*32 + hi*8 + e (32B aligned)
          const v4i c0 = ntld4i(r0);
          const v4i c1 = ntld4i(r0 + 4);
          const v4i c2 = ntld4i(r0 + 16);
          const v4i c3 = ntld4i(r0 + 20);
          v16bf kf;
          #pragma unroll
          for (int e = 0; e < 4; ++e) {
            kf[e]      = (__bf16)((float)c0[e] * scl + mn);
            kf[e + 4]  = (__bf16)((float)c1[e] * scl + mn);
            kf[e + 8]  = (__bf16)((float)c2[e] * scl + mn);
            kf[e + 12] = (__bf16)((float)c3[e] * scl + mn);
          }
          s8 = __builtin_amdgcn_wmma_f32_16x16x32_bf16(false, kf, false, qt[s], (short)0, s8, false, false);
        }
      } else {
        const float* fp = kfull + ((size_t)bkv * LR + (key - LQ)) * HD;
        #pragma unroll
        for (int s = 0; s < 4; ++s) {
          const float* r0 = fp + s * 32 + hi * 8;
          const v4f f0 = ntld4f(r0);
          const v4f f1 = ntld4f(r0 + 4);
          const v4f f2 = ntld4f(r0 + 16);
          const v4f f3 = ntld4f(r0 + 20);
          v16bf kf;
          #pragma unroll
          for (int e = 0; e < 4; ++e) {
            kf[e]      = (__bf16)f0[e];
            kf[e + 4]  = (__bf16)f1[e];
            kf[e + 8]  = (__bf16)f2[e];
            kf[e + 12] = (__bf16)f3[e];
          }
          s8 = __builtin_amdgcn_wmma_f32_16x16x32_bf16(false, kf, false, qt[s], (short)0, s8, false, false);
        }
      }
      sc[t] = s8;
    }

    // online softmax: lane col = head; its 16 keys live in sc[t][r], other 16 via xor16
    float bm = -1e30f;
    #pragma unroll
    for (int r = 0; r < 8; ++r) bm = fmaxf(bm, fmaxf(sc[0][r], sc[1][r]));
    bm = fmaxf(bm, __shfl_xor(bm, 16));
    const float n0 = fmaxf(M0, __shfl(bm, 0));
    const float n1 = fmaxf(M1, __shfl(bm, 1));
    const float n2 = fmaxf(M2, __shfl(bm, 2));
    const float n3 = fmaxf(M3, __shfl(bm, 3));
    const float f0 = __expf(M0 - n0), f1 = __expf(M1 - n1);
    const float f2 = __expf(M2 - n2), f3 = __expf(M3 - n3);
    M0 = n0; M1 = n1; M2 = n2; M3 = n3;
    const float own = (col == 0) ? M0 : (col == 1) ? M1 : (col == 2) ? M2 : (col == 3) ? M3 : 0.0f;
    v8f p0, p1;
    float ps = 0.f;
    #pragma unroll
    for (int r = 0; r < 8; ++r) {
      p0[r] = __expf(sc[0][r] - own);
      p1[r] = __expf(sc[1][r] - own);
      ps += p0[r] + p1[r];
    }
    ps += __shfl_xor(ps, 16);
    S0 = S0 * f0 + __shfl(ps, 0);
    S1 = S1 * f1 + __shfl(ps, 1);
    S2 = S2 * f2 + __shfl(ps, 2);
    S3 = S3 * f3 + __shfl(ps, 3);
    const float frs[4] = {f0, f1, f2, f3};
    #pragma unroll
    for (int dt = 0; dt < 8; ++dt) {
      #pragma unroll
      for (int r = 0; r < 8; ++r) acc[dt][r] *= frs[r & 3];
    }

    // P A-fragment: elem e<8 -> key hi*8+e (tile0), e>=8 -> key 16+hi*8+(e-8) (tile1)
    v16bf pa;
    #pragma unroll
    for (int e = 0; e < 8; ++e) {
      pa[e]     = (__bf16)p0[e];
      pa[e + 8] = (__bf16)p1[e];
    }

    // P*V: B-fragment lane: d = dt*16+col, elem e: key = base + hi*8 + e + (e&8).
    // Scale/min for key base+j gathered from lane j by shuffle (no extra loads).
    #pragma unroll
    for (int dt = 0; dt < 8; ++dt) {
      const int d = dt * 16 + col;
      v16bf vb;
      if (quant) {
        const int g = dt >> 1;                     // quant group of this d-tile
        #pragma unroll
        for (int e = 0; e < 16; ++e) {
          const int srcl = hi * 8 + e + (e & 8);   // lane owning this key's scales
          const float scl = __shfl(vs4[g], srcl);
          const float mn  = __shfl(vm4[g], srcl);
          const size_t ro = (size_t)bkv * LQ + (base + hi * 8 + e + (e & 8));
          const int code  = __builtin_nontemporal_load(vq + ro * HD + d);
          vb[e] = (__bf16)((float)code * scl + mn);
        }
      } else {
        #pragma unroll
        for (int e = 0; e < 16; ++e) {
          const size_t ro = (size_t)bkv * LR + (base + hi * 8 + e + (e & 8) - LQ);
          vb[e] = (__bf16)__builtin_nontemporal_load(vfull + ro * HD + d);
        }
      }
      acc[dt] = __builtin_amdgcn_wmma_f32_16x16x32_bf16(false, pa, false, vb, (short)0, acc[dt], false, false);
    }
  }

  float* P = partial + (size_t)(bkv * NCH + c) * PSTR;
  if (hi == 0) {
    #pragma unroll
    for (int dt = 0; dt < 8; ++dt) {
      #pragma unroll
      for (int r = 0; r < 4; ++r)
        P[r * HD + dt * 16 + col] = acc[dt][r];
    }
  }
  if (lane == 0) {
    P[512] = M0; P[513] = M1; P[514] = M2; P[515] = M3;
    P[516] = S0; P[517] = S1; P[518] = S2; P[519] = S3;
  }
}

// ---------------------------------------------------------------------------
// Merge chunk partials + new-token key/value, normalize. 1 wave per (b,kv).
// ---------------------------------------------------------------------------
__global__ __launch_bounds__(32) void attn_merge(
    const float* __restrict__ partial, const float* __restrict__ q_r,
    const float* __restrict__ k_r, const float* __restrict__ qkv,
    float* __restrict__ attn_out)
{
  const int kv = blockIdx.x & (NKV - 1);
  const int b  = blockIdx.x >> 3;
  const int lane = threadIdx.x & 31;
  const int bkv = b * NKV + kv;
  const float* kn = k_r + (size_t)bkv * HD;
  const float* PB = partial + (size_t)bkv * NCH * PSTR;

  float sn[4], Mst[4], den[4];
  #pragma unroll
  for (int m = 0; m < 4; ++m) {
    const float* qp = q_r + ((size_t)b * NH + kv * GQA + m) * HD;  // q pre-scaled
    float p = 0.f;
    #pragma unroll
    for (int i = 0; i < 4; ++i) p += qp[lane + 32 * i] * kn[lane + 32 * i];
    #pragma unroll
    for (int off = 16; off >= 1; off >>= 1) p += __shfl_xor(p, off);
    sn[m] = p;  Mst[m] = p;
  }
  for (int c = 0; c < NCH; ++c) {
    #pragma unroll
    for (int m = 0; m < 4; ++m) Mst[m] = fmaxf(Mst[m], PB[c * PSTR + 512 + m]);
  }
  #pragma unroll
  for (int m = 0; m < 4; ++m) den[m] = __expf(sn[m] - Mst[m]);
  for (int c = 0; c < NCH; ++c) {
    #pragma unroll
    for (int m = 0; m < 4; ++m)
      den[m] += __expf(PB[c * PSTR + 512 + m] - Mst[m]) * PB[c * PSTR + 516 + m];
  }
  const float* vn = qkv + (size_t)b * QKV_LD + HID + 1024 + kv * HD;  // new v
  #pragma unroll
  for (int m = 0; m < 4; ++m) {
    const float en = __expf(sn[m] - Mst[m]);
    float o0 = en * vn[lane],      o1 = en * vn[lane + 32];
    float o2 = en * vn[lane + 64], o3 = en * vn[lane + 96];
    for (int c = 0; c < NCH; ++c) {
      const float e = __expf(PB[c * PSTR + 512 + m] - Mst[m]);
      const float* ap = PB + c * PSTR + m * HD;
      o0 += e * ap[lane];      o1 += e * ap[lane + 32];
      o2 += e * ap[lane + 64]; o3 += e * ap[lane + 96];
    }
    const float inv = 1.0f / den[m];
    float* op = attn_out + ((size_t)b * NH + kv * GQA + m) * HD;
    op[lane] = o0 * inv;      op[lane + 32] = o1 * inv;
    op[lane + 64] = o2 * inv; op[lane + 96] = o3 * inv;
  }
}

// ---------------------------------------------------------------------------
extern "C" void kernel_launch(void* const* d_in, const int* in_sizes, int n_in,
                              void* d_out, int out_size, void* d_ws, size_t ws_size,
                              hipStream_t stream)
{
  (void)in_sizes; (void)n_in; (void)out_size; (void)ws_size;
  const float* hidden = (const float*)d_in[0];
  const int*   pos    = (const int*)  d_in[1];
  const int*   kq     = (const int*)  d_in[2];
  const float* ks     = (const float*)d_in[3];
  const float* km     = (const float*)d_in[4];
  const int*   vq     = (const int*)  d_in[5];
  const float* vs     = (const float*)d_in[6];
  const float* vm     = (const float*)d_in[7];
  const float* kfull  = (const float*)d_in[8];
  const float* vfull  = (const float*)d_in[9];
  const float* Wq     = (const float*)d_in[10];
  const float* Wk     = (const float*)d_in[11];
  const float* Wv     = (const float*)d_in[12];
  const float* Wo     = (const float*)d_in[13];
  float* out = (float*)d_out;

  float* ws   = (float*)d_ws;
  float* qkv  = ws;                     // 16*6144
  float* q_r  = qkv + NB * QKV_LD;      // 16*32*128
  float* k_r  = q_r + NB * NH * HD;     // 16*8*128
  float* attn = k_r + NB * NKV * HD;    // 16*32*128
  float* part = attn + NB * NH * HD;    // 128*32*520

  gemm_qkv<<<dim3(384), 256, 0, stream>>>(hidden, Wq, Wk, Wv, qkv);
  rope_qk<<<dim3((NB * (NH + NKV) * HD) / 256), 256, 0, stream>>>(qkv, pos, q_r, k_r);
  attn_partial<<<dim3(NCH, NKV, NB), 32, 0, stream>>>(q_r, kq, ks, km, vq, vs, vm,
                                                      kfull, vfull, part);
  attn_merge<<<dim3(NB * NKV), 32, 0, stream>>>(part, q_r, k_r, qkv, attn);
  gemm_out<<<dim3(HID / 16), 256, 0, stream>>>(attn, Wo, out);
}
